// _EGNNLayer_48455821033952
// MI455X (gfx1250) — matrix operand order repaired and checked by reading, not compiled, not run on hardware
//
#include <hip/hip_runtime.h>
#include <hip/hip_bf16.h>

// ---------------------------------------------------------------------------
// EGNN layer for MI455X (gfx1250, wave32, WMMA f16->f32).
// All GEMMs run on v_wmma_f32_16x16x32_f16; activations converted to f16,
// accumulation in f32. B-fragment loads are software-pipelined (depth 2) so
// L2 latency is hidden behind the WMMA pipe. Scatter aggregation uses native
// f32 global atomics (L2-resident, 192MB L2 >> 25.6MB targets).
// ---------------------------------------------------------------------------

typedef _Float16 v16h __attribute__((ext_vector_type(16)));
typedef _Float16 v8h  __attribute__((ext_vector_type(8)));
typedef float    v8f  __attribute__((ext_vector_type(8)));

union AFrag { v16h v; v8h h[2]; };

__device__ __forceinline__ float silu(float x) {
    return x / (1.0f + __expf(-x));
}

__device__ __forceinline__ void atomAddF(float* p, float v) {
    unsafeAtomicAdd(p, v);   // -> global_atomic_add_f32
}

// ---------------------------------------------------------------------------
// GEMM over 8 N-tiles x KT K-tiles with depth-2 software pipelining of the
// B-fragment loads (each is 32B contiguous, L2-resident weight pack).
// ---------------------------------------------------------------------------
template <int KT>
__device__ __forceinline__ void gemm_tiles(const AFrag* __restrict__ a,
                                           const _Float16* __restrict__ Wp,
                                           int lane, v8f* __restrict__ acc) {
    constexpr int TOT = 8 * KT;
    const _Float16* bp = Wp + lane * 16;
    auto loadB = [&](int t) -> v16h {
        int nt = t / KT, kt = t - nt * KT;
        return *(const v16h*)(bp + (kt * 8 + nt) * 512);
    };
    v16h bbuf0 = loadB(0);
    v16h bbuf1 = (TOT > 1) ? loadB(1) : bbuf0;
    #pragma unroll
    for (int t = 0; t < TOT; ++t) {
        const int nt = t / KT, kt = t - nt * KT;
        v16h bcur = (t & 1) ? bbuf1 : bbuf0;
        // prefetch two steps ahead while the WMMA below executes
        if (t + 2 < TOT) {
            if (t & 1) bbuf1 = loadB(t + 2);
            else       bbuf0 = loadB(t + 2);
        }
        acc[nt] = __builtin_amdgcn_wmma_f32_16x16x32_f16(
            false, a[kt].v, false, bcur, (short)0, acc[nt], false, false);
    }
}

// ---------------------------------------------------------------------------
// Pack a row-major [K x 128] f32 weight into WMMA-B f16 fragments:
// dst[((kt*8+nt)*32 + lane)*16 + e]  where K = kt*32 + (lane&16 ? 8:0) + (e&7) + (e&8 ? 16:0),
// N = nt*16 + (lane&15).  K zero-padded to KT*32.
// ---------------------------------------------------------------------------
__global__ void pack_w(const float* __restrict__ W, _Float16* __restrict__ dst,
                       int Kreal, int KT) {
    int idx = blockIdx.x * 256 + threadIdx.x;
    int total = KT * 8 * 32 * 16;
    if (idx >= total) return;
    int e    = idx & 15;
    int lane = (idx >> 4) & 31;
    int nt   = (idx >> 9) & 7;
    int kt   = idx >> 12;
    int k = kt * 32 + ((lane & 16) ? 8 : 0) + (e & 7) + ((e & 8) ? 16 : 0);
    int n = nt * 16 + (lane & 15);
    float v = (k < Kreal) ? W[k * 128 + n] : 0.0f;
    dst[idx] = (_Float16)v;
}

// film[g][c] = b_film[c] + sum_k cond[g][k] * W_film[k][c]   (G x 256)
__global__ void film_kernel(const float* __restrict__ cond,
                            const float* __restrict__ Wf,
                            const float* __restrict__ bf,
                            float* __restrict__ film, int G) {
    int idx = blockIdx.x * 256 + threadIdx.x;
    if (idx >= G * 256) return;
    int g = idx >> 8, c = idx & 255;
    float s = bf[c];
    #pragma unroll 8
    for (int k = 0; k < 128; ++k) s += cond[g * 128 + k] * Wf[k * 256 + c];
    film[idx] = s;
}

// ---------------------------------------------------------------------------
// Edge kernel: 128 threads (4 waves), 64 edges / block, 16 edges / wave.
// LDS layout (54784 B):
//   msgin  f16 [64][288]  @0       (36864)   -- overlaid by c1out later
//   c1out  f32 [64][129]  @0       (33024)   -- after msgin is dead
//   mij    f16 [64][128]  @36864   (16384)
//   idxi   i32 [64]       @53248
//   idxj   i32 [64]       @53504
//   rel    f32 [3][64]    @53760
//   d2a    f32 [64]       @54528
// ---------------------------------------------------------------------------
__global__ __launch_bounds__(128)
void edge_kernel(const float* __restrict__ h, const float* __restrict__ pos,
                 const float* __restrict__ eattr, const int* __restrict__ ei,
                 const _Float16* __restrict__ Wp1, const float* __restrict__ b1,
                 const _Float16* __restrict__ Wp2, const float* __restrict__ b2,
                 const _Float16* __restrict__ Wpc1, const float* __restrict__ bc1,
                 const float* __restrict__ Wc2,
                 float* __restrict__ agg_msg, float* __restrict__ agg_coord,
                 int N, int E) {
    __shared__ __align__(16) char smem[54784];
    _Float16* msgin = (_Float16*)smem;
    float*    c1out = (float*)smem;
    _Float16* mij   = (_Float16*)(smem + 36864);
    int*      idxi  = (int*)(smem + 53248);
    int*      idxj  = (int*)(smem + 53504);
    float*    rel   = (float*)(smem + 53760);
    float*    d2a   = (float*)(smem + 54528);

    const int tid  = threadIdx.x;
    const int lane = tid & 31;
    const int wave = tid >> 5;
    const int base = blockIdx.x * 64;

    // L2 warm hint for the weight packs
    __builtin_prefetch(Wp1 + (size_t)tid * 64, 0, 1);
    __builtin_prefetch(Wp2 + (size_t)tid * 32, 0, 1);

    // ---- indices, rel_pos, dist^2, edge_attr, zero-pad --------------------
    if (tid < 64) {
        int ge = base + tid; if (ge >= E) ge = E - 1;
        int ni = ei[ge], nj = ei[E + ge];
        idxi[tid] = ni; idxj[tid] = nj;
        float rx = pos[ni * 3 + 0] - pos[nj * 3 + 0];
        float ry = pos[ni * 3 + 1] - pos[nj * 3 + 1];
        float rz = pos[ni * 3 + 2] - pos[nj * 3 + 2];
        float dd = rx * rx + ry * ry + rz * rz;
        rel[tid] = rx; rel[64 + tid] = ry; rel[128 + tid] = rz;
        d2a[tid] = dd;
        msgin[tid * 288 + 256] = (_Float16)dd;
    }
    for (int t = tid; t < 64 * 16; t += 128) {           // edge_attr cols 257..272
        int e = t >> 4, a = t & 15;
        int ge = base + e; if (ge >= E) ge = E - 1;
        msgin[e * 288 + 257 + a] = (_Float16)eattr[ge * 16 + a];
    }
    for (int t = tid; t < 64 * 15; t += 128) {           // pad cols 273..287
        int e = t / 15, a = t - e * 15;
        msgin[e * 288 + 273 + a] = (_Float16)0.0f;
    }
    __syncthreads();

    // ---- gather h[i], h[j] (L2-resident random rows) ----------------------
    {
        const int col = tid;                              // 0..127
        for (int e = 0; e < 64; ++e) {
            int ni = idxi[e], nj = idxj[e];
            msgin[e * 288 + col]       = (_Float16)h[(size_t)ni * 128 + col];
            msgin[e * 288 + 128 + col] = (_Float16)h[(size_t)nj * 128 + col];
        }
    }
    __syncthreads();

    const int mrow   = wave * 16 + (lane & 15);           // tile row this lane owns (A)
    const int khi    = (lane & 16) ? 8 : 0;               // A layout: hi lanes offset K+8
    const int ncol   = lane & 15;                         // C/D column within N-tile
    const int mbase2 = (lane & 16) ? 8 : 0;               // C/D row base

    // ---- msg1: [64x288(f16)] x [288x128] ----------------------------------
    AFrag a1[9];
    #pragma unroll
    for (int kt = 0; kt < 9; ++kt) {
        const _Float16* p = msgin + mrow * 288 + kt * 32 + khi;
        a1[kt].h[0] = *(const v8h*)p;
        a1[kt].h[1] = *(const v8h*)(p + 16);
    }
    __syncthreads();                                      // msgin dead -> c1out overlay OK

    v8f acc[8];
    #pragma unroll
    for (int nt = 0; nt < 8; ++nt) { v8f z = {}; acc[nt] = z; }
    gemm_tiles<9>(a1, Wp1, lane, acc);

    #pragma unroll
    for (int nt = 0; nt < 8; ++nt) {                      // bias + SiLU -> m tile
        int n = nt * 16 + ncol;
        float bn = b1[n];
        #pragma unroll
        for (int r = 0; r < 8; ++r) {
            float s = silu(acc[nt][r] + bn);
            mij[(wave * 16 + mbase2 + r) * 128 + n] = (_Float16)s;
        }
    }

    // ---- msg2: [64x128] x [128x128] ---------------------------------------
    AFrag a2[4];
    #pragma unroll
    for (int kt = 0; kt < 4; ++kt) {
        const _Float16* p = mij + mrow * 128 + kt * 32 + khi;
        a2[kt].h[0] = *(const v8h*)p;
        a2[kt].h[1] = *(const v8h*)(p + 16);
    }
    #pragma unroll
    for (int nt = 0; nt < 8; ++nt) { v8f z = {}; acc[nt] = z; }
    gemm_tiles<4>(a2, Wp2, lane, acc);

    #pragma unroll
    for (int nt = 0; nt < 8; ++nt) {                      // m_ij: SiLU, scatter, store f16
        int n = nt * 16 + ncol;
        float bn = b2[n];
        #pragma unroll
        for (int r = 0; r < 8; ++r) {
            int m = wave * 16 + mbase2 + r;
            float s = silu(acc[nt][r] + bn);
            mij[m * 128 + n] = (_Float16)s;
            if (base + m < E)
                atomAddF(agg_msg + (size_t)idxi[m] * 128 + n, s);
        }
    }

    // ---- coord mlp stage 1: [64x128] x [128x128] --------------------------
    AFrag a3[4];
    #pragma unroll
    for (int kt = 0; kt < 4; ++kt) {
        const _Float16* p = mij + mrow * 128 + kt * 32 + khi;
        a3[kt].h[0] = *(const v8h*)p;
        a3[kt].h[1] = *(const v8h*)(p + 16);
    }
    #pragma unroll
    for (int nt = 0; nt < 8; ++nt) { v8f z = {}; acc[nt] = z; }
    gemm_tiles<4>(a3, Wpc1, lane, acc);

    #pragma unroll
    for (int nt = 0; nt < 8; ++nt) {                      // SiLU -> c1out (f32, stride 129)
        int n = nt * 16 + ncol;
        float bn = bc1[n];
        #pragma unroll
        for (int r = 0; r < 8; ++r) {
            float s = silu(acc[nt][r] + bn);
            c1out[(wave * 16 + mbase2 + r) * 129 + n] = s;
        }
    }
    __syncthreads();

    // ---- coord weight dot + normalized scatter ----------------------------
    if (tid < 64) {
        int ge = base + tid;
        if (ge < E) {
            const float* row = c1out + tid * 129;
            float s = 0.0f;
            #pragma unroll 16
            for (int c = 0; c < 128; ++c) s += row[c] * Wc2[c];
            float inv = rsqrtf(d2a[tid] + 1e-8f);
            int node = idxi[tid];
            atomAddF(agg_coord + (size_t)node * 3 + 0, rel[tid]       * inv * s);
            atomAddF(agg_coord + (size_t)node * 3 + 1, rel[64 + tid]  * inv * s);
            atomAddF(agg_coord + (size_t)node * 3 + 2, rel[128 + tid] * inv * s);
        }
    }
}

// ---------------------------------------------------------------------------
// Node kernel: 128 threads (4 waves), 64 nodes / block.
// LDS (49920 B):
//   xin  f16 [64][256] @0      (32768)  -- overlaid by X later
//   u    f16 [64][128] @32768  (16384)
//   X    f32 [64][129] @0      (33024)  -- after xin+u row0 dead (barrier)
//   blds i32 [64]      @49152
//   mu   f32 [64]      @49408
//   rstd f32 [64]      @49664
// ---------------------------------------------------------------------------
__global__ __launch_bounds__(128)
void node_kernel(const float* __restrict__ h, const float* __restrict__ pos,
                 const int* __restrict__ batch,
                 const _Float16* __restrict__ Wpn1, const float* __restrict__ bn1,
                 const _Float16* __restrict__ Wpn2, const float* __restrict__ bn2,
                 const float* __restrict__ film,
                 const float* __restrict__ lng, const float* __restrict__ lnb,
                 const float* __restrict__ agg_msg, const float* __restrict__ agg_coord,
                 float* __restrict__ out, int N) {
    __shared__ __align__(16) char smem[49920];
    _Float16* xin  = (_Float16*)smem;
    _Float16* u    = (_Float16*)(smem + 32768);
    float*    X    = (float*)smem;
    int*      blds = (int*)(smem + 49152);
    float*    muL  = (float*)(smem + 49408);
    float*    rsL  = (float*)(smem + 49664);

    const int tid  = threadIdx.x;
    const int lane = tid & 31;
    const int wave = tid >> 5;
    const int base = blockIdx.x * 64;

    __builtin_prefetch(Wpn1 + (size_t)tid * 64, 0, 1);
    __builtin_prefetch(Wpn2 + (size_t)tid * 32, 0, 1);

    // ---- gather [h | agg_msg] -> f16 tile ---------------------------------
    if (tid < 64) {
        int row = base + tid; if (row >= N) row = N - 1;
        blds[tid] = batch[row];
    }
    for (int e = 0; e < 64; ++e) {
        int row = base + e; if (row >= N) row = N - 1;
        xin[e * 256 + tid]       = (_Float16)h[(size_t)row * 128 + tid];
        xin[e * 256 + 128 + tid] = (_Float16)agg_msg[(size_t)row * 128 + tid];
    }
    __syncthreads();

    const int mrow   = wave * 16 + (lane & 15);
    const int khi    = (lane & 16) ? 8 : 0;
    const int ncol   = lane & 15;
    const int mbase2 = (lane & 16) ? 8 : 0;

    // ---- n1: [64x256] x [256x128] -----------------------------------------
    AFrag a1[8];
    #pragma unroll
    for (int kt = 0; kt < 8; ++kt) {
        const _Float16* p = xin + mrow * 256 + kt * 32 + khi;
        a1[kt].h[0] = *(const v8h*)p;
        a1[kt].h[1] = *(const v8h*)(p + 16);
    }
    __syncthreads();                                      // xin dead

    v8f acc[8];
    #pragma unroll
    for (int nt = 0; nt < 8; ++nt) { v8f z = {}; acc[nt] = z; }
    gemm_tiles<8>(a1, Wpn1, lane, acc);

    #pragma unroll
    for (int nt = 0; nt < 8; ++nt) {                      // bias + SiLU -> u
        int n = nt * 16 + ncol;
        float bn = bn1[n];
        #pragma unroll
        for (int r = 0; r < 8; ++r) {
            float s = silu(acc[nt][r] + bn);
            u[(wave * 16 + mbase2 + r) * 128 + n] = (_Float16)s;
        }
    }

    // ---- n2: [64x128] x [128x128] -----------------------------------------
    AFrag a2[4];
    #pragma unroll
    for (int kt = 0; kt < 4; ++kt) {
        const _Float16* p = u + mrow * 128 + kt * 32 + khi;
        a2[kt].h[0] = *(const v8h*)p;
        a2[kt].h[1] = *(const v8h*)(p + 16);
    }
    __syncthreads();                                      // u dead -> X overlay fully safe

    #pragma unroll
    for (int nt = 0; nt < 8; ++nt) { v8f z = {}; acc[nt] = z; }
    gemm_tiles<4>(a2, Wpn2, lane, acc);

    #pragma unroll
    for (int nt = 0; nt < 8; ++nt) {                      // hn -> X (f32, stride 129)
        int n = nt * 16 + ncol;
        float bn = bn2[n];
        #pragma unroll
        for (int r = 0; r < 8; ++r)
            X[(wave * 16 + mbase2 + r) * 129 + n] = acc[nt][r] + bn;
    }
    __syncthreads();

    // ---- FiLM + residual: x = h + gamma*hn + beta -------------------------
    for (int t = tid; t < 64 * 128; t += 128) {
        int e = t >> 7, c = t & 127;
        int row = base + e;
        if (row < N) {
            int g = blds[e];
            float hn = X[e * 129 + c];
            X[e * 129 + c] = h[(size_t)row * 128 + c]
                           + film[g * 256 + c] * hn + film[g * 256 + 128 + c];
        }
    }
    __syncthreads();

    // ---- LayerNorm stats (thread per row, stride-129 avoids conflicts) ----
    if (tid < 64) {
        float s = 0.0f, s2 = 0.0f;
        const float* row = X + tid * 129;
        #pragma unroll 16
        for (int c = 0; c < 128; ++c) { float v = row[c]; s += v; s2 += v * v; }
        float mu = s * (1.0f / 128.0f);
        float var = s2 * (1.0f / 128.0f) - mu * mu;
        muL[tid] = mu;
        rsL[tid] = rsqrtf(var + 1e-5f);
    }
    __syncthreads();

    // ---- final normalized write (coalesced) + pos_new ---------------------
    for (int t = tid; t < 64 * 128; t += 128) {
        int e = t >> 7, c = t & 127;
        int row = base + e;
        if (row < N)
            out[(size_t)row * 128 + c] =
                (X[e * 129 + c] - muL[e]) * rsL[e] * lng[c] + lnb[c];
    }
    float* pout = out + (size_t)N * 128;
    for (int t = tid; t < 64 * 3; t += 128) {
        int e = t / 3, k = t - e * 3;
        int row = base + e;
        if (row < N)
            pout[(size_t)row * 3 + k] = pos[(size_t)row * 3 + k]
                                      + agg_coord[(size_t)row * 3 + k];
    }
}

// ---------------------------------------------------------------------------
// Host launcher
// ---------------------------------------------------------------------------
extern "C" void kernel_launch(void* const* d_in, const int* in_sizes, int n_in,
                              void* d_out, int out_size, void* d_ws, size_t ws_size,
                              hipStream_t stream) {
    const float* h     = (const float*)d_in[0];
    const float* pos   = (const float*)d_in[1];
    const float* eattr = (const float*)d_in[2];
    const float* cond  = (const float*)d_in[3];
    const int*   ei    = (const int*)d_in[4];
    const int*   batch = (const int*)d_in[5];
    const float* Wm1   = (const float*)d_in[6];
    const float* bm1   = (const float*)d_in[7];
    const float* Wm2   = (const float*)d_in[8];
    const float* bm2   = (const float*)d_in[9];
    const float* Wc1   = (const float*)d_in[10];
    const float* bc1   = (const float*)d_in[11];
    const float* Wc2   = (const float*)d_in[12];
    const float* Wn1   = (const float*)d_in[13];
    const float* bn1   = (const float*)d_in[14];
    const float* Wn2   = (const float*)d_in[15];
    const float* bn2   = (const float*)d_in[16];
    const float* Wfilm = (const float*)d_in[17];
    const float* bfilm = (const float*)d_in[18];
    const float* lng   = (const float*)d_in[19];
    const float* lnb   = (const float*)d_in[20];

    const int N = in_sizes[0] / 128;
    const int E = in_sizes[2] / 16;
    const int G = in_sizes[3] / 128;

    char* ws = (char*)d_ws;
    _Float16* W1p  = (_Float16*)(ws + 0);        // 9*8*512 f16  = 73728 B
    _Float16* W2p  = (_Float16*)(ws + 73728);    // 4*8*512 f16  = 32768 B
    _Float16* Wc1p = (_Float16*)(ws + 106496);   // 32768 B
    _Float16* Wn1p = (_Float16*)(ws + 139264);   // 8*8*512 f16  = 65536 B
    _Float16* Wn2p = (_Float16*)(ws + 204800);   // 32768 B
    float*    filmp = (float*)(ws + 237568);     // G*256*4
    size_t offAggM = 237568 + (size_t)G * 256 * 4;
    offAggM = (offAggM + 255) & ~(size_t)255;
    float* agg_msg   = (float*)(ws + offAggM);                       // N*128*4
    float* agg_coord = (float*)(ws + offAggM + (size_t)N * 128 * 4); // N*3*4

    // zero scatter targets (stream-ordered, capture-legal)
    hipMemsetAsync(agg_msg,   0, (size_t)N * 128 * 4, stream);
    hipMemsetAsync(agg_coord, 0, (size_t)N * 3 * 4,   stream);

    // pack weights into WMMA-B fragment layout (f16)
    auto packLaunch = [&](const float* W, _Float16* dst, int Kreal, int KT) {
        int total = KT * 8 * 32 * 16;
        pack_w<<<(total + 255) / 256, 256, 0, stream>>>(W, dst, Kreal, KT);
    };
    packLaunch(Wm1, W1p, 273, 9);
    packLaunch(Wm2, W2p, 128, 4);
    packLaunch(Wc1, Wc1p, 128, 4);
    packLaunch(Wn1, Wn1p, 256, 8);
    packLaunch(Wn2, Wn2p, 128, 4);

    film_kernel<<<(G * 256 + 255) / 256, 256, 0, stream>>>(cond, Wfilm, bfilm, filmp, G);

    edge_kernel<<<(E + 63) / 64, 128, 0, stream>>>(
        h, pos, eattr, ei, W1p, bm1, W2p, bm2, Wc1p, bc1, Wc2,
        agg_msg, agg_coord, N, E);

    node_kernel<<<(N + 63) / 64, 128, 0, stream>>>(
        h, pos, batch, Wn1p, bn1, Wn2p, bn2, filmp, lng, lnb,
        agg_msg, agg_coord, (float*)d_out, N);
}
